// ContinuousThoughtMachine_56659208569014
// MI455X (gfx1250) — matrix operand order repaired and checked by reading, not compiled, not run on hardware
//
#include <hip/hip_runtime.h>
#include <math.h>

#define B_      16
#define DMODEL  2048
#define DIN     512
#define MEM_    25
#define NOUT    256
#define NACT    128
#define OUTD    1000
#define SZA     8256
#define SZO     32896
#define ITERS_  15

typedef __attribute__((ext_vector_type(16))) _Float16 v16h;
typedef __attribute__((ext_vector_type(8)))  _Float16 v8h;
typedef __attribute__((ext_vector_type(8)))  float    v8f;

// ------------------------------------------------------------------ utils
__device__ __forceinline__ float sigmoidf_(float x){ return 1.f/(1.f+expf(-x)); }

__device__ __forceinline__ float blk_sum(float v, float* red){
  int t = threadIdx.x;
  red[t] = v; __syncthreads();
  for (int s = blockDim.x>>1; s>0; s>>=1){ if (t<s) red[t]+=red[t+s]; __syncthreads(); }
  float r = red[0]; __syncthreads(); return r;
}
__device__ __forceinline__ float blk_max(float v, float* red){
  int t = threadIdx.x;
  red[t] = v; __syncthreads();
  for (int s = blockDim.x>>1; s>0; s>>=1){ if (t<s) red[t]=fmaxf(red[t],red[t+s]); __syncthreads(); }
  float r = red[0]; __syncthreads(); return r;
}

__global__ void k_fill_f32(float* p, float v, long n){
  long i = (long)blockIdx.x*blockDim.x + threadIdx.x;
  if (i < n) p[i] = v;
}
__global__ void k_fill_f16(_Float16* p, long n){
  long i = (long)blockIdx.x*blockDim.x + threadIdx.x;
  if (i < n) p[i] = (_Float16)0.f;
}
__global__ void k_f32_to_f16(_Float16* d, const float* s, long n){
  long i = (long)blockIdx.x*blockDim.x + threadIdx.x;
  if (i < n) d[i] = (_Float16)s[i];
}
// src f32 [K][N] row-major -> dst f16 [N][K]
__global__ void k_transpose_f16(_Float16* __restrict__ dst, const float* __restrict__ src,
                                int K, int N){
  __shared__ float tile[32][33];
  int kb = blockIdx.x*32, nb = blockIdx.y*32;
  for (int i = threadIdx.y; i < 32; i += blockDim.y){
    int k = kb + i, n = nb + threadIdx.x;
    tile[i][threadIdx.x] = (k < K && n < N) ? src[(size_t)k*N + n] : 0.f;
  }
  __syncthreads();
  for (int i = threadIdx.y; i < 32; i += blockDim.y){
    int n = nb + i, k = kb + threadIdx.x;
    if (n < N && k < K) dst[(size_t)n*K + k] = (_Float16)tile[threadIdx.x][i];
  }
}

// ------------------------------------------------------------- WMMA GEMMs
__device__ __forceinline__ v16h load_afrag(const _Float16* arow, int kb, int half){
  v8h r0 = *(const v8h*)(arow + kb + half*16);
  v8h r1 = *(const v8h*)(arow + kb + half*16 + 8);
  v16h a;
#pragma unroll
  for (int e = 0; e < 8; ++e){ a[e]=r0[e]; a[e+8]=r1[e]; }
  return a;
}

// C[M,N] = A[M,K](f16 row-major) * Bt[N,K](f16) + bias.
// One wave computes a 16x64 tile: A fragment reused across 4 N-tiles.
// OOB N columns stream from zrow (zero-filled dummy) -> no exec divergence in loop.
__global__ void k_gemm_wmma(float* __restrict__ C, const _Float16* __restrict__ A,
                            const _Float16* __restrict__ Bt, const float* __restrict__ bias,
                            const _Float16* __restrict__ zrow, int M, int N, int K){
  const int lane = threadIdx.x, half = lane>>4, l16 = lane&15;
  const int mb = blockIdx.y*16, nb0 = blockIdx.x*64;
  const _Float16* arow = A + (size_t)(mb + l16)*K;
  const _Float16* brow[4];
  bool bv[4];
#pragma unroll
  for (int t = 0; t < 4; ++t){
    int n = nb0 + t*16 + l16;
    bv[t] = (n < N);
    brow[t] = bv[t] ? (Bt + (size_t)n*K) : zrow;
  }
  v8f acc[4] = {};
  for (int kb = 0; kb < K; kb += 32){
    v16h a = load_afrag(arow, kb, half);
#pragma unroll
    for (int t = 0; t < 4; ++t){
      v16h bf = *(const v16h*)(brow[t] + kb + half*16);
      acc[t] = __builtin_amdgcn_wmma_f32_16x16x32_f16(false, a, false, bf, (short)0, acc[t], false, false);
    }
  }
#pragma unroll
  for (int t = 0; t < 4; ++t){
    if (!bv[t]) continue;
    int n = nb0 + t*16 + l16;
    float bvv = bias ? bias[n] : 0.f;
#pragma unroll
    for (int r = 0; r < 8; ++r)
      C[(size_t)(mb + half*8 + r)*N + n] = acc[t][r] + bvv;
  }
}

// split-K over grid.z, atomicAdd into zeroed C; bias added by z==0 blocks. 16x64 tile.
__global__ void k_gemm_wmma_splitk(float* __restrict__ C, const _Float16* __restrict__ A,
                                   const _Float16* __restrict__ Bt, const float* __restrict__ bias,
                                   const _Float16* __restrict__ zrow, int M, int N, int K){
  const int lane = threadIdx.x, half = lane>>4, l16 = lane&15;
  const int mb = blockIdx.y*16, nb0 = blockIdx.x*64;
  const _Float16* arow = A + (size_t)(mb + l16)*K;
  const _Float16* brow[4];
  bool bv[4];
#pragma unroll
  for (int t = 0; t < 4; ++t){
    int n = nb0 + t*16 + l16;
    bv[t] = (n < N);
    brow[t] = bv[t] ? (Bt + (size_t)n*K) : zrow;
  }
  v8f acc[4] = {};
  const int ksteps = K >> 5;
  for (int s = blockIdx.z; s < ksteps; s += gridDim.z){
    const int kb = s << 5;
    v16h a = load_afrag(arow, kb, half);
#pragma unroll
    for (int t = 0; t < 4; ++t){
      v16h bf = *(const v16h*)(brow[t] + kb + half*16);
      acc[t] = __builtin_amdgcn_wmma_f32_16x16x32_f16(false, a, false, bf, (short)0, acc[t], false, false);
    }
  }
#pragma unroll
  for (int t = 0; t < 4; ++t){
    if (!bv[t]) continue;
    int n = nb0 + t*16 + l16;
    float bvv = (bias && blockIdx.z == 0) ? bias[n] : 0.f;
#pragma unroll
    for (int r = 0; r < 8; ++r)
      atomicAdd(&C[(size_t)(mb + half*8 + r)*N + n], acc[t][r] + bvv);
  }
}

// conv2 implicit-GEMM: A = im2col(hpad f16 (B,34,34,512)), Bt[512][4608].
// One wave computes a 64x16 tile: weight fragment reused across 4 M-tiles.
__global__ void k_conv2_gemm(float* __restrict__ C, const _Float16* __restrict__ hpad,
                             const _Float16* __restrict__ Wt, const float* __restrict__ bias){
  const int lane = threadIdx.x, half = lane>>4, l16 = lane&15;
  const int mb0 = blockIdx.y*64, nb = blockIdx.x*16;
  const int n = nb + l16;
  const _Float16* wrow = Wt + (size_t)n*4608;
  int bb[4], yy[4], xx[4];
#pragma unroll
  for (int t = 0; t < 4; ++t){
    int m = mb0 + t*16 + l16;
    bb[t] = m >> 10; int rem = m & 1023; yy[t] = rem >> 5; xx[t] = rem & 31;
  }
  v8f acc[4] = {};
  for (int kb = 0; kb < 4608; kb += 32){
    const int k0 = kb + half*16, k1 = k0 + 8;
    const int ci0 = k0 & 511, t0 = k0 >> 9, kx0 = t0 % 3, ky0 = t0 / 3;
    const int ci1 = k1 & 511, t1 = k1 >> 9, kx1 = t1 % 3, ky1 = t1 / 3;
    v16h bf = *(const v16h*)(wrow + kb + half*16);
#pragma unroll
    for (int t = 0; t < 4; ++t){
      v8h r0 = *(const v8h*)(hpad + (((size_t)bb[t]*34 + (yy[t]+ky0))*34 + (xx[t]+kx0))*512 + ci0);
      v8h r1 = *(const v8h*)(hpad + (((size_t)bb[t]*34 + (yy[t]+ky1))*34 + (xx[t]+kx1))*512 + ci1);
      v16h a;
#pragma unroll
      for (int e = 0; e < 8; ++e){ a[e]=r0[e]; a[e+8]=r1[e]; }
      acc[t] = __builtin_amdgcn_wmma_f32_16x16x32_f16(false, a, false, bf, (short)0, acc[t], false, false);
    }
  }
  float bvv = bias[n];
#pragma unroll
  for (int t = 0; t < 4; ++t)
#pragma unroll
    for (int r = 0; r < 8; ++r)
      C[(size_t)(mb0 + t*16 + half*8 + r)*512 + n] = acc[t][r] + bvv;
}

// --------------------------------------------------------- conv1 / pooling
__global__ void k_conv1(_Float16* __restrict__ hpad, const float* __restrict__ x,
                        const float* __restrict__ w, const float* __restrict__ cb,
                        const float* __restrict__ bs, const float* __restrict__ bb,
                        const float* __restrict__ bm, const float* __restrict__ bv){
  const int blk = blockIdx.x;
  const int b = blk >> 10, rem = blk & 1023, py = rem >> 5, px = rem & 31;
  __shared__ float patch[4][4][3];
  const int t = threadIdx.x;
  if (t < 48){
    int ci = t % 3, pxx = (t/3) % 4, pyy = t/12;
    int iy = 2*py - 1 + pyy, ix = 2*px - 1 + pxx;
    float v = 0.f;
    if (iy >= 0 && iy < 64 && ix >= 0 && ix < 64)
      v = x[(((size_t)b*64 + iy)*64 + ix)*3 + ci];
    patch[pyy][pxx][ci] = v;
  }
  __syncthreads();
  for (int co = t; co < 512; co += 128){
    float inv = rsqrtf(bv[co] + 1e-5f);
    float sc = bs[co], bc = bb[co], mc = bm[co], c0 = cb[co];
    float best = -INFINITY;
    for (int dy = 0; dy < 2; ++dy)
    for (int dx = 0; dx < 2; ++dx){
      float acc = c0;
      for (int ky = 0; ky < 3; ++ky)
      for (int kx = 0; kx < 3; ++kx)
      for (int ci = 0; ci < 3; ++ci)
        acc += patch[dy+ky][dx+kx][ci] * w[((ky*3 + kx)*3 + ci)*512 + co];
      best = fmaxf(best, fmaxf((acc - mc)*inv*sc + bc, 0.f));
    }
    hpad[(((size_t)b*34 + (py+1))*34 + (px+1))*512 + co] = (_Float16)best;
  }
}

__global__ void k_bn2_pool(_Float16* __restrict__ out16, const float* __restrict__ craw,
                           const float* __restrict__ s, const float* __restrict__ bb,
                           const float* __restrict__ m, const float* __restrict__ v){
  const int blk = blockIdx.x;
  const int b = blk >> 8, rem = blk & 255, py = rem >> 4, px = rem & 15;
  for (int c = threadIdx.x; c < 512; c += blockDim.x){
    float inv = rsqrtf(v[c] + 1e-5f);
    float sc = s[c], bc = bb[c], mc = m[c];
    float best = -INFINITY;
    for (int dy = 0; dy < 2; ++dy)
    for (int dx = 0; dx < 2; ++dx){
      float val = craw[(((size_t)b*32 + 2*py+dy)*32 + 2*px+dx)*512 + c];
      best = fmaxf(best, fmaxf((val - mc)*inv*sc + bc, 0.f));
    }
    out16[((size_t)b*256 + rem)*512 + c] = (_Float16)best;
  }
}

// --------------------------------------------------------------- LN / GLU
__global__ void k_ln_f16(_Float16* __restrict__ out, const float* __restrict__ in,
                         const float* __restrict__ s, const float* __restrict__ b, int D){
  __shared__ float red[256];
  const int row = blockIdx.x, t = threadIdx.x;
  const float* x = in + (size_t)row*D;
  float sum = 0.f, sq = 0.f;
  for (int i = t; i < D; i += 256){ float v = x[i]; sum += v; sq += v*v; }
  sum = blk_sum(sum, red); sq = blk_sum(sq, red);
  float mu = sum / D, var = sq / D - mu*mu;
  float inv = rsqrtf(var + 1e-6f);
  _Float16* o = out + (size_t)row*D;
  for (int i = t; i < D; i += 256) o[i] = (_Float16)((x[i]-mu)*inv*s[i] + b[i]);
}

__global__ void k_glu_ln(float* __restrict__ pre, const float* __restrict__ syn,
                         const float* __restrict__ s, const float* __restrict__ b){
  __shared__ float g[2048];
  __shared__ float red[256];
  const int bi = blockIdx.x, t = threadIdx.x;
  const float* row = syn + (size_t)bi*4096;
  for (int i = t; i < 2048; i += 256) g[i] = row[i] * sigmoidf_(row[i+2048]);
  __syncthreads();
  float sum = 0.f, sq = 0.f;
  for (int i = t; i < 2048; i += 256){ float v = g[i]; sum += v; sq += v*v; }
  sum = blk_sum(sum, red); sq = blk_sum(sq, red);
  float mu = sum / 2048.f, var = sq / 2048.f - mu*mu;
  float inv = rsqrtf(var + 1e-6f);
  for (int i = t; i < 2048; i += 256)
    pre[(size_t)bi*2048 + i] = (g[i]-mu)*inv*s[i] + b[i];
}

// ----------------------------------------------------------------- sync
__global__ void k_sync(_Float16* __restrict__ sy16, float* __restrict__ sy32,
                       float* __restrict__ alpha, float* __restrict__ beta,
                       const float* __restrict__ act, const float* __restrict__ dec,
                       int nsel, int offset, int sz){
  int p = blockIdx.x*blockDim.x + threadIdx.x;
  if (p >= sz) return;
  float nf = (float)nsel;
  int i = (int)((2.f*nf + 1.f - sqrtf((2.f*nf+1.f)*(2.f*nf+1.f) - 8.f*(float)p)) * 0.5f);
  if (i < 0) i = 0; if (i >= nsel) i = nsel-1;
  while (i > 0 && i*(2*nsel - i + 1)/2 > p) --i;
  while ((i+1)*(2*nsel - i)/2 <= p) ++i;
  int j = i + (p - i*(2*nsel - i + 1)/2);
  float r = expf(-dec[p]);
  for (int b = 0; b < B_; ++b){
    const float* arow = act + (size_t)b*DMODEL + offset;
    float na = r*alpha[(size_t)b*sz + p] + arow[i]*arow[j];
    float nb = r*beta [(size_t)b*sz + p] + 1.f;
    alpha[(size_t)b*sz + p] = na; beta[(size_t)b*sz + p] = nb;
    float sy = na * rsqrtf(nb);
    sy16[(size_t)b*sz + p] = (_Float16)sy;
    if (sy32) sy32[(size_t)b*sz + p] = sy;
  }
}

// ------------------------------------------------------------- attention
__global__ void k_attn(_Float16* __restrict__ out16, const float* __restrict__ qh,
                       const float* __restrict__ kh, const float* __restrict__ vh){
  const int b = blockIdx.x >> 3, h = blockIdx.x & 7;
  __shared__ float w[256];
  __shared__ float red[256];
  __shared__ float qs[64];
  const int t = threadIdx.x;
  if (t < 64) qs[t] = qh[(size_t)b*512 + h*64 + t];
  __syncthreads();
  const float* krow = kh + ((size_t)b*256 + t)*512 + h*64;
  float sc = 0.f;
  for (int d = 0; d < 64; ++d) sc += qs[d]*krow[d];
  sc *= 0.125f;
  float mx = blk_max(sc, red);
  float e = expf(sc - mx);
  float ssum = blk_sum(e, red);
  w[t] = e / ssum;
  __syncthreads();
  if (t < 64){
    float acc = 0.f;
    for (int s = 0; s < 256; ++s) acc += w[s] * vh[((size_t)b*256 + s)*512 + h*64 + t];
    out16[(size_t)b*512 + h*64 + t] = (_Float16)acc;
  }
}

__global__ void k_cat(_Float16* __restrict__ cat, const float* __restrict__ attno,
                      const float* __restrict__ act){
  int idx = blockIdx.x*blockDim.x + threadIdx.x;
  if (idx >= B_*2560) return;
  int b = idx / 2560, c = idx % 2560;
  float v = (c < 512) ? attno[(size_t)b*512 + c] : act[(size_t)b*2048 + (c-512)];
  cat[idx] = (_Float16)v;
}

// ------------------------------------------- per-neuron SuperLinear stack
__global__ void k_superlinear(float* __restrict__ act, float* __restrict__ trace,
                              const float* __restrict__ pre,
                              const float* __restrict__ w1, const float* __restrict__ b1,
                              const float* __restrict__ w2, const float* __restrict__ b2){
  const int n = blockIdx.x, t = threadIdx.x;  // 128 threads
  __shared__ float tr[16][25];
  __shared__ float h[16][128];
  __shared__ float hg[16][64];
  __shared__ float out2[16][2];
  for (int idx = t; idx < 16*25; idx += 128){
    int b = idx / 25, m = idx % 25;
    tr[b][m] = (m < 24) ? trace[((size_t)b*DMODEL + n)*25 + m + 1] : pre[(size_t)b*DMODEL + n];
  }
  __syncthreads();
  for (int idx = t; idx < 16*25; idx += 128){
    int b = idx / 25, m = idx % 25;
    trace[((size_t)b*DMODEL + n)*25 + m] = tr[b][m];
  }
  float wv[25];
#pragma unroll
  for (int m = 0; m < 25; ++m) wv[m] = w1[((size_t)m*128 + t)*DMODEL + n];
  float bias1 = b1[(size_t)n*128 + t];
  for (int b = 0; b < 16; ++b){
    float acc = bias1;
#pragma unroll
    for (int m = 0; m < 25; ++m) acc += tr[b][m]*wv[m];
    h[b][t] = acc;
  }
  __syncthreads();
  if (t < 64)
    for (int b = 0; b < 16; ++b) hg[b][t] = h[b][t] * sigmoidf_(h[b][t+64]);
  __syncthreads();
  if (t < 32){
    int b = t >> 1, c = t & 1;
    float acc = b2[(size_t)n*2 + c];
    for (int o = 0; o < 64; ++o) acc += hg[b][o] * w2[((size_t)o*2 + c)*DMODEL + n];
    out2[b][c] = acc;
  }
  __syncthreads();
  if (t < 16) act[(size_t)t*DMODEL + n] = out2[t][0] * sigmoidf_(out2[t][1]);
}

// ---------------------------------------------------- entropy + outputs
__global__ void k_out(float* __restrict__ dout, const float* __restrict__ logits, int iter){
  __shared__ float red[256];
  const int b = blockIdx.x, t = threadIdx.x;
  const float* z = logits + (size_t)b*OUTD;
  float mx = -INFINITY;
  for (int i = t; i < OUTD; i += 256) mx = fmaxf(mx, z[i]);
  mx = blk_max(mx, red);
  float s = 0.f, tt = 0.f;
  for (int i = t; i < OUTD; i += 256){ float u = z[i]-mx; float e = expf(u); s += e; tt += e*u; }
  s = blk_sum(s, red); tt = blk_sum(tt, red);
  float ent = logf(s) - tt/s;
  float ne = ent / logf((float)OUTD);
  for (int i = t; i < OUTD; i += 256)
    dout[((size_t)b*OUTD + i)*ITERS_ + iter] = z[i];
  if (t == 0){
    dout[(size_t)B_*OUTD*ITERS_ + b*2*ITERS_ + iter] = ne;
    dout[(size_t)B_*OUTD*ITERS_ + b*2*ITERS_ + ITERS_ + iter] = 1.f - ne;
  }
}

__global__ void k_bcast_act(float* act, const float* ss){
  int i = blockIdx.x*blockDim.x + threadIdx.x;
  if (i < B_*DMODEL) act[i] = ss[i % DMODEL];
}
__global__ void k_bcast_trace(float* trace, const float* st){
  long i = (long)blockIdx.x*blockDim.x + threadIdx.x;
  if (i < (long)B_*DMODEL*MEM_) trace[i] = st[i % (DMODEL*MEM_)];
}

// ======================================================================
extern "C" void kernel_launch(void* const* d_in, const int* in_sizes, int n_in,
                              void* d_out, int out_size, void* d_ws, size_t ws_size,
                              hipStream_t stream){
  (void)in_sizes; (void)n_in; (void)out_size; (void)ws_size;
  const float* x       = (const float*)d_in[0];
  const float* conv1_w = (const float*)d_in[1];  const float* conv1_b = (const float*)d_in[2];
  const float* bn1_s = (const float*)d_in[3], *bn1_b = (const float*)d_in[4],
             * bn1_m = (const float*)d_in[5], *bn1_v = (const float*)d_in[6];
  const float* conv2_w = (const float*)d_in[7];  const float* conv2_b = (const float*)d_in[8];
  const float* bn2_s = (const float*)d_in[9], *bn2_b = (const float*)d_in[10],
             * bn2_m = (const float*)d_in[11], *bn2_v = (const float*)d_in[12];
  const float* kv_w = (const float*)d_in[13], *kv_b = (const float*)d_in[14];
  const float* kvln_s = (const float*)d_in[15], *kvln_b = (const float*)d_in[16];
  const float* q_w = (const float*)d_in[17], *q_b = (const float*)d_in[18];
  const float* aq_w = (const float*)d_in[19], *aq_b = (const float*)d_in[20];
  const float* ak_w = (const float*)d_in[21], *ak_b = (const float*)d_in[22];
  const float* av_w = (const float*)d_in[23], *av_b = (const float*)d_in[24];
  const float* ao_w = (const float*)d_in[25], *ao_b = (const float*)d_in[26];
  const float* syn_w = (const float*)d_in[27], *syn_b = (const float*)d_in[28];
  const float* synln_s = (const float*)d_in[29], *synln_b = (const float*)d_in[30];
  const float* sl1_w = (const float*)d_in[31], *sl1_b = (const float*)d_in[32];
  const float* sl2_w = (const float*)d_in[33], *sl2_b = (const float*)d_in[34];
  const float* outp_w = (const float*)d_in[35], *outp_b = (const float*)d_in[36];
  const float* start_state = (const float*)d_in[37];
  const float* start_trace = (const float*)d_in[38];
  const float* dec_act = (const float*)d_in[39];
  const float* dec_out = (const float*)d_in[40];
  float* dout = (float*)d_out;

  // ---------------- workspace carve ----------------
  char* wp = (char*)d_ws;
  auto carve = [&](size_t bytes) -> void* {
    void* p = (void*)wp; wp += (bytes + 255) & ~(size_t)255; return p;
  };
  _Float16* hpad    = (_Float16*)carve((size_t)B_*34*34*512*2);
  float*    c2raw   = (float*)   carve((size_t)B_*32*32*512*4);
  _Float16* h2p16   = (_Float16*)carve((size_t)B_*256*512*2);
  float*    kvraw   = (float*)   carve((size_t)B_*256*512*4);
  _Float16* kv16    = (_Float16*)carve((size_t)B_*256*512*2);
  float*    kh      = (float*)   carve((size_t)B_*256*512*4);
  float*    vh      = (float*)   carve((size_t)B_*256*512*4);
  _Float16* conv2_wt= (_Float16*)carve((size_t)512*4608*2);
  _Float16* kv_wt   = (_Float16*)carve((size_t)512*512*2);
  _Float16* q_wt    = (_Float16*)carve((size_t)512*SZA*2);
  _Float16* aq_wt   = (_Float16*)carve((size_t)512*512*2);
  _Float16* ak_wt   = (_Float16*)carve((size_t)512*512*2);
  _Float16* av_wt   = (_Float16*)carve((size_t)512*512*2);
  _Float16* ao_wt   = (_Float16*)carve((size_t)512*512*2);
  _Float16* syn_wt  = (_Float16*)carve((size_t)4096*2560*2);
  _Float16* outp_wt = (_Float16*)carve((size_t)OUTD*SZO*2);
  _Float16* zrow    = (_Float16*)carve((size_t)SZO*2);          // zero dummy B row (max K)
  float*    act     = (float*)   carve((size_t)B_*DMODEL*4);
  float*    trace   = (float*)   carve((size_t)B_*DMODEL*MEM_*4);
  float*    aA      = (float*)   carve((size_t)B_*SZA*4);
  float*    bA      = (float*)   carve((size_t)B_*SZA*4);
  float*    aO      = (float*)   carve((size_t)B_*SZO*4);
  float*    bO      = (float*)   carve((size_t)B_*SZO*4);
  _Float16* syA16   = (_Float16*)carve((size_t)B_*SZA*2);
  _Float16* syO16   = (_Float16*)carve((size_t)B_*SZO*2);
  float*    qbuf    = (float*)   carve((size_t)B_*512*4);
  _Float16* q16     = (_Float16*)carve((size_t)B_*512*2);
  float*    qhb     = (float*)   carve((size_t)B_*512*4);
  _Float16* atth    = (_Float16*)carve((size_t)B_*512*2);
  float*    attno   = (float*)   carve((size_t)B_*512*4);
  _Float16* cat     = (_Float16*)carve((size_t)B_*2560*2);
  float*    synout  = (float*)   carve((size_t)B_*4096*4);
  float*    pre     = (float*)   carve((size_t)B_*DMODEL*4);
  float*    logits  = (float*)   carve((size_t)B_*OUTD*4);

  dim3 tb(32, 8);
  auto tgrid = [](int K, int N){ return dim3((K+31)/32, (N+31)/32); };
  // weight conversion (transposed f16), every call (no cached state)
  k_transpose_f16<<<tgrid(4608,512), tb, 0, stream>>>(conv2_wt, conv2_w, 4608, 512);
  k_transpose_f16<<<tgrid(512,512),  tb, 0, stream>>>(kv_wt, kv_w, 512, 512);
  k_transpose_f16<<<tgrid(SZA,512),  tb, 0, stream>>>(q_wt,  q_w,  SZA, 512);
  k_transpose_f16<<<tgrid(512,512),  tb, 0, stream>>>(aq_wt, aq_w, 512, 512);
  k_transpose_f16<<<tgrid(512,512),  tb, 0, stream>>>(ak_wt, ak_w, 512, 512);
  k_transpose_f16<<<tgrid(512,512),  tb, 0, stream>>>(av_wt, av_w, 512, 512);
  k_transpose_f16<<<tgrid(2560,4096),tb, 0, stream>>>(syn_wt, syn_w, 2560, 4096);
  k_transpose_f16<<<tgrid(512,512),  tb, 0, stream>>>(ao_wt, ao_w, 512, 512);
  k_transpose_f16<<<tgrid(SZO,OUTD), tb, 0, stream>>>(outp_wt, outp_w, SZO, OUTD);
  k_fill_f16<<<(SZO+255)/256, 256, 0, stream>>>(zrow, (long)SZO);

  // backbone
  long nhpad = (long)B_*34*34*512;
  k_fill_f16<<<(int)((nhpad+255)/256), 256, 0, stream>>>(hpad, nhpad);
  k_conv1<<<B_*32*32, 128, 0, stream>>>(hpad, x, conv1_w, conv1_b, bn1_s, bn1_b, bn1_m, bn1_v);
  k_conv2_gemm<<<dim3(32, 256), 32, 0, stream>>>(c2raw, hpad, conv2_wt, conv2_b);
  k_bn2_pool<<<B_*256, 256, 0, stream>>>(h2p16, c2raw, bn2_s, bn2_b, bn2_m, bn2_v);
  k_gemm_wmma<<<dim3(8, 256), 32, 0, stream>>>(kvraw, h2p16, kv_wt, kv_b, zrow, 4096, 512, 512);
  k_ln_f16<<<4096, 256, 0, stream>>>(kv16, kvraw, kvln_s, kvln_b, 512);
  k_gemm_wmma<<<dim3(8, 256), 32, 0, stream>>>(kh, kv16, ak_wt, ak_b, zrow, 4096, 512, 512);
  k_gemm_wmma<<<dim3(8, 256), 32, 0, stream>>>(vh, kv16, av_wt, av_b, zrow, 4096, 512, 512);

  // recurrent state init
  k_bcast_act<<<(B_*DMODEL+255)/256, 256, 0, stream>>>(act, start_state);
  k_bcast_trace<<<(B_*DMODEL*MEM_+255)/256, 256, 0, stream>>>(trace, start_trace);
  k_fill_f32<<<(B_*SZA+255)/256, 256, 0, stream>>>(aA, 0.f, (long)B_*SZA);
  k_fill_f32<<<(B_*SZA+255)/256, 256, 0, stream>>>(bA, 0.f, (long)B_*SZA);
  k_fill_f32<<<(B_*SZO+255)/256, 256, 0, stream>>>(aO, 0.f, (long)B_*SZO);
  k_fill_f32<<<(B_*SZO+255)/256, 256, 0, stream>>>(bO, 0.f, (long)B_*SZO);

  float* syO_out = dout + (size_t)B_*OUTD*ITERS_ + (size_t)B_*2*ITERS_;

  for (int it = 0; it < ITERS_; ++it){
    // action synchronisation -> syA (f16)
    k_sync<<<(SZA+255)/256, 256, 0, stream>>>(syA16, (float*)nullptr, aA, bA, act,
                                              dec_act, NACT, DMODEL-NACT, SZA);
    // q = syA @ q_w + q_b   (split-K over K=8256)
    k_fill_f32<<<(B_*512+255)/256, 256, 0, stream>>>(qbuf, 0.f, (long)B_*512);
    k_gemm_wmma_splitk<<<dim3(8, 1, 8), 32, 0, stream>>>(qbuf, syA16, q_wt, q_b, zrow, 16, 512, SZA);
    k_f32_to_f16<<<(B_*512+255)/256, 256, 0, stream>>>(q16, qbuf, (long)B_*512);
    // qh = q @ aq_w + aq_b
    k_gemm_wmma<<<dim3(8, 1), 32, 0, stream>>>(qhb, q16, aq_wt, aq_b, zrow, 16, 512, 512);
    // attention heads
    k_attn<<<B_*8, 256, 0, stream>>>(atth, qhb, kh, vh);
    // attn_out = attn @ ao_w + ao_b
    k_gemm_wmma<<<dim3(8, 1), 32, 0, stream>>>(attno, atth, ao_wt, ao_b, zrow, 16, 512, 512);
    // synapse: cat -> GEMM (split-K) -> GLU -> LN
    k_cat<<<(B_*2560+255)/256, 256, 0, stream>>>(cat, attno, act);
    k_fill_f32<<<(B_*4096+255)/256, 256, 0, stream>>>(synout, 0.f, (long)B_*4096);
    k_gemm_wmma_splitk<<<dim3(64, 1, 4), 32, 0, stream>>>(synout, cat, syn_wt, syn_b, zrow, 16, 4096, 2560);
    k_glu_ln<<<B_, 256, 0, stream>>>(pre, synout, synln_s, synln_b);
    // trace shift + SuperLinear stack -> new act
    k_superlinear<<<DMODEL, 128, 0, stream>>>(act, trace, pre, sl1_w, sl1_b, sl2_w, sl2_b);
    // output synchronisation -> syO (f16 + f32 straight into d_out)
    k_sync<<<(SZO+255)/256, 256, 0, stream>>>(syO16, syO_out, aO, bO, act, dec_out, NOUT, 0, SZO);
    // logits = syO @ outp_w + outp_b  (split-K over K=32896, N=1000 padded to 1024)
    k_fill_f32<<<(B_*OUTD+255)/256, 256, 0, stream>>>(logits, 0.f, (long)B_*OUTD);
    k_gemm_wmma_splitk<<<dim3(16, 1, 16), 32, 0, stream>>>(logits, syO16, outp_wt, outp_b, zrow, 16, OUTD, SZO);
    // entropy + predictions/certainties
    k_out<<<B_, 256, 0, stream>>>(dout, logits, it);
  }
}
